// SupConLoss_79869211836801
// MI455X (gfx1250) — matrix operand order repaired and checked by reading, not compiled, not run on hardware
//
#include <hip/hip_runtime.h>

#define N_TOT   4096
#define B_ROWS  2048
#define DFEAT   128
#define CDIM    128
#define INV_T   (1.0f / 0.07f)
#define ALPHA_C 0.7f

typedef __bf16 bf16_t;
typedef bf16_t v16bf __attribute__((ext_vector_type(16)));
typedef float  v8f   __attribute__((ext_vector_type(8)));

// trivially-constructible container for a WMMA 16-bf16 operand (32 bytes)
union V16 {
    v16bf    v;
    unsigned u[8];
};

static __device__ __forceinline__ void ld16(void* dst, const void* src) {
    *(uint4*)dst = *(const uint4*)src;
}

static __device__ __forceinline__ unsigned short f2bf(float f) {
    union { float f; unsigned u; } c;
    c.f = f;
    unsigned u = c.u;
    u += 0x7FFFu + ((u >> 16) & 1u);   // round-to-nearest-even
    return (unsigned short)(u >> 16);
}
static __device__ __forceinline__ float bf2f(unsigned short h) {
    union { unsigned u; float f; } c;
    c.u = ((unsigned)h) << 16;
    return c.f;
}
static __device__ __forceinline__ v8f vzero8() {
    v8f z;
#pragma unroll
    for (int i = 0; i < 8; ++i) z[i] = 0.0f;
    return z;
}

// ---------------------------------------------------------------------------
// Kernel 1: view-major gather of features -> bf16 matrix Xc[4096][128] in ws,
// plus per-row logit max m_i = (sum_k bf16(x_ik)^2) / T  (diagonal of G).
// One wave per row; 8 waves / block.
// ---------------------------------------------------------------------------
__global__ __launch_bounds__(256) void k_prep_x(const float* __restrict__ feats,
                                                unsigned short* __restrict__ xbf,
                                                float* __restrict__ mrow) {
    const int wid = threadIdx.x >> 5, lane = threadIdx.x & 31;
    const int i = blockIdx.x * 8 + wid;                 // contrast row, 0..4095
    const int b = i & (B_ROWS - 1), vview = i >> 11;    // concat([f[:,0], f[:,1]])
    const float4 x = *(const float4*)(feats + (unsigned)(b * 2 + vview) * DFEAT + lane * 4);
    const unsigned short h0 = f2bf(x.x), h1 = f2bf(x.y), h2 = f2bf(x.z), h3 = f2bf(x.w);
    uint2 pk;
    pk.x = (unsigned)h0 | ((unsigned)h1 << 16);
    pk.y = (unsigned)h2 | ((unsigned)h3 << 16);
    *(uint2*)(xbf + (unsigned)i * DFEAT + lane * 4) = pk;
    const float f0 = bf2f(h0), f1 = bf2f(h1), f2 = bf2f(h2), f3 = bf2f(h3);
    float ss = f0 * f0 + f1 * f1 + f2 * f2 + f3 * f3;
#pragma unroll
    for (int off = 1; off <= 16; off <<= 1) ss += __shfl_xor(ss, off, 32);
    if (lane == 0) mrow[i] = ss * INV_T;
}

// ---------------------------------------------------------------------------
// Kernel 2: D2^T (bf16) = transpose(concat(D, D)) -> [128][4096] in ws, so the
// B operand of the second WMMA (E @ D2) is two 16B loads per lane from one d2t row.
// ---------------------------------------------------------------------------
__global__ __launch_bounds__(256) void k_prep_d2t(const float* __restrict__ Dm,
                                                  unsigned short* __restrict__ d2t) {
    const int idx = blockIdx.x * 256 + threadIdx.x;     // 0 .. 128*4096-1
    const int j = idx & (N_TOT - 1);
    const int c = idx >> 12;
    d2t[(unsigned)c * N_TOT + j] = f2bf(Dm[(unsigned)(j & (B_ROWS - 1)) * CDIM + c]);
}

// ---------------------------------------------------------------------------
// Kernel 3 (main fused pass): one block per 16-row strip; the 4 waves of the
// block split the j dimension (1024 columns each) -> 1024 independent waves.
// Per wave, per 32-wide j chunk:
//   G tiles      : 8x v_wmma_f32_16x16x32_bf16 (A strip resident in VGPRs)
//   elementwise  : logits, exp(+l)/exp(-l), mask stats (all in registers)
//   E relayout   : C-layout -> A-layout via 1.25KB per-wave LDS tile
//   E @ D2       : 8x v_wmma_f32_16x16x32_bf16 accumulating [16 x 128]
// Epilogue: partials combined through LDS; rows distributed across waves for
// the D_hat softmax and the per-row loss term.
// ---------------------------------------------------------------------------
__global__ __launch_bounds__(128) void k_main(const unsigned short* __restrict__ xbf,
                                              const unsigned short* __restrict__ d2t,
                                              const float* __restrict__ mrow,
                                              const float* __restrict__ maskM,
                                              const float* __restrict__ Dm,
                                              const float* __restrict__ Lm,
                                              float* __restrict__ dout,
                                              float* __restrict__ rowloss) {
    const int lane = threadIdx.x & 31;
    const int wid  = threadIdx.x >> 5;
    const int hf   = lane >> 4;        // half-wave select
    const int lrow = lane & 15;
    const int i0   = blockIdx.x * 16;  // strip base row
    const int jbeg = wid * (N_TOT / 4);
    const int jend = jbeg + (N_TOT / 4);

    // per-wave E staging tile: 16 rows x 32 cols bf16, row stride 40 (80B, 16B aligned)
    __shared__ __align__(16) unsigned short ldsE[4][16 * 40];
    __shared__ __align__(16) float part[4][16][CDIM];   // per-wave E@D2 partials
    __shared__ float statS[4][16], statZ[4][16], statP[4][16], statM[4][16];
    unsigned short* eT = ldsE[wid];

    // A strip (rows i0..i0+15, K=128 as 4 chunks of 32) in WMMA A-layout:
    // lanes 0-15: row=lane, K {0..7,16..23}; lanes 16-31: row=lane-16, K {8..15,24..31}
    V16 aA[4];
    {
        const unsigned rowA = (unsigned)(i0 + lrow) * DFEAT;
#pragma unroll
        for (int kc = 0; kc < 4; ++kc) {
            const unsigned k1 = rowA + kc * 32 + (hf ? 8 : 0);
            ld16(&aA[kc].u[0], xbf + k1);
            ld16(&aA[kc].u[4], xbf + k1 + 16);
        }
    }

    float mr[8];
#pragma unroll
    for (int r = 0; r < 8; ++r) mr[r] = mrow[i0 + r + hf * 8];

    float sAcc[8], zAcc[8], pAcc[8], msAcc[8];
#pragma unroll
    for (int r = 0; r < 8; ++r) { sAcc[r] = 0.f; zAcc[r] = 0.f; pAcc[r] = 0.f; msAcc[r] = 0.f; }
    v8f acc[8];
#pragma unroll
    for (int t = 0; t < 8; ++t) acc[t] = vzero8();

    for (int j0 = jbeg; j0 < jend; j0 += 32) {
        if (j0 + 32 < jend) {
            __builtin_prefetch(xbf + (unsigned)(j0 + 32 + lrow) * DFEAT, 0, 1);
            __builtin_prefetch(xbf + (unsigned)(j0 + 48 + lrow) * DFEAT, 0, 1);
        }
        // ---- first matmul: two 16x16 G tiles (cols j0..+15, j0+16..+31)
        v8f g0 = vzero8(), g1 = vzero8();
        {
            const unsigned bp0 = (unsigned)(j0 + lrow) * DFEAT;
            const unsigned bp1 = (unsigned)(j0 + 16 + lrow) * DFEAT;
#pragma unroll
            for (int kc = 0; kc < 4; ++kc) {
                const unsigned k0 = kc * 32 + (hf ? 8 : 0);   // B: K {0..7,16..23} / {8..15,24..31} per half
                V16 b0, b1;
                ld16(&b0.u[0], xbf + bp0 + k0);  ld16(&b0.u[4], xbf + bp0 + k0 + 16);
                ld16(&b1.u[0], xbf + bp1 + k0);  ld16(&b1.u[4], xbf + bp1 + k0 + 16);
                g0 = __builtin_amdgcn_wmma_f32_16x16x32_bf16(false, aA[kc].v, false, b0.v,
                                                             (short)0, g0, false, false);
                g1 = __builtin_amdgcn_wmma_f32_16x16x32_bf16(false, aA[kc].v, false, b1.v,
                                                             (short)0, g1, false, false);
            }
        }
        // ---- elementwise: logits, exps, mask stats; stage E (bf16) into LDS
#pragma unroll
        for (int jt = 0; jt < 2; ++jt) {
            const int jcol = j0 + jt * 16 + lrow;
#pragma unroll
            for (int r = 0; r < 8; ++r) {
                const int irow = i0 + r + hf * 8;         // C-layout row mapping
                const float g  = (jt ? g1[r] : g0[r]);
                const float l  = g * INV_T - mr[r];
                const float offd = (irow != jcol) ? 1.0f : 0.0f;
                const float mk   = maskM[(unsigned)irow * N_TOT + jcol] * offd;
                zAcc[r]  += __expf(l)  * offd;
                const float eneg = __expf(-l) * offd;
                sAcc[r]  += eneg;
                pAcc[r]   = __fmaf_rn(mk, l, pAcc[r]);
                msAcc[r] += mk;
                eT[(r + hf * 8) * 40 + jt * 16 + lrow] = f2bf(eneg);
            }
        }
        asm volatile("s_wait_dscnt 0x0" ::: "memory");     // same-wave LDS RAW
        // ---- E tile in A-layout (M=i rows, K=32 j values)
        V16 a2;
        {
            const unsigned short* tp = eT + lrow * 40 + (hf ? 8 : 0);
            ld16(&a2.u[0], tp);
            ld16(&a2.u[4], tp + 16);
        }
        // ---- second matmul: E[16x32] @ D2[32x128] accumulated over j
#pragma unroll
        for (int t = 0; t < 8; ++t) {
            const unsigned dp = (unsigned)(t * 16 + lrow) * N_TOT + j0 + (hf ? 8 : 0);
            V16 b2;
            ld16(&b2.u[0], d2t + dp);  ld16(&b2.u[4], d2t + dp + 16);
            acc[t] = __builtin_amdgcn_wmma_f32_16x16x32_bf16(false, a2.v, false, b2.v,
                                                             (short)0, acc[t], false, false);
        }
    }

    // ---- row-stat butterflies within each half-wave (rows match C-layout halves)
#pragma unroll
    for (int r = 0; r < 8; ++r) {
#pragma unroll
        for (int off = 1; off <= 8; off <<= 1) {
            sAcc[r]  += __shfl_xor(sAcc[r],  off, 32);
            zAcc[r]  += __shfl_xor(zAcc[r],  off, 32);
            pAcc[r]  += __shfl_xor(pAcc[r],  off, 32);
            msAcc[r] += __shfl_xor(msAcc[r], off, 32);
        }
    }
    // ---- publish this wave's partials (stats + E@D2 tile) to LDS
#pragma unroll
    for (int r = 0; r < 8; ++r) {
        if (lrow == 0) {
            const int row = r + hf * 8;
            statS[wid][row] = sAcc[r];
            statZ[wid][row] = zAcc[r];
            statP[wid][row] = pAcc[r];
            statM[wid][row] = msAcc[r];
        }
    }
#pragma unroll
    for (int t = 0; t < 8; ++t) {
#pragma unroll
        for (int r = 0; r < 8; ++r) {
            part[wid][r + hf * 8][t * 16 + lrow] = acc[t][r];
        }
    }
    __syncthreads();

    // ---- epilogue: each wave finishes 4 complete rows (softmax over C=128)
#pragma unroll
    for (int q = 0; q < 4; ++q) {
        const int row  = wid * 4 + q;
        const int irow = i0 + row;
        float s = 0.f, z = 0.f, p = 0.f, ms = 0.f;
#pragma unroll
        for (int w = 0; w < 4; ++w) {
            s  += statS[w][row];
            z  += statZ[w][row];
            p  += statP[w][row];
            ms += statM[w][row];
        }
        const float invs = 1.0f / s;
        float v[4];
#pragma unroll
        for (int cc = 0; cc < 4; ++cc) {
            const int c = cc * 32 + lane;
            float e = 0.f;
#pragma unroll
            for (int w = 0; w < 4; ++w) e += part[w][row][c];
            const unsigned src = (unsigned)(irow & (B_ROWS - 1)) * CDIM + c;
            v[cc] = ALPHA_C * (Dm[src] - e * invs) + (1.0f - ALPHA_C) * Lm[src];
        }
        float mx = fmaxf(fmaxf(v[0], v[1]), fmaxf(v[2], v[3]));
#pragma unroll
        for (int off = 1; off <= 16; off <<= 1) mx = fmaxf(mx, __shfl_xor(mx, off, 32));
        float sum = 0.f;
#pragma unroll
        for (int cc = 0; cc < 4; ++cc) {
            v[cc] = __expf(v[cc] - mx);
            sum += v[cc];
        }
#pragma unroll
        for (int off = 1; off <= 16; off <<= 1) sum += __shfl_xor(sum, off, 32);
        const float invsum = 1.0f / sum;
#pragma unroll
        for (int cc = 0; cc < 4; ++cc)
            dout[1 + (unsigned)irow * CDIM + cc * 32 + lane] = v[cc] * invsum;
        if (lane == 0)
            rowloss[irow] = -(p - ms * __logf(z)) / (ms + 1e-5f);
    }
}

// ---------------------------------------------------------------------------
// Kernel 4: deterministic loss reduction (no atomics -> replay-stable)
// ---------------------------------------------------------------------------
__global__ __launch_bounds__(256) void k_loss(const float* __restrict__ rowloss,
                                              float* __restrict__ dout) {
    __shared__ float red[256];
    float s = 0.0f;
    for (int i = threadIdx.x; i < N_TOT; i += 256) s += rowloss[i];
    red[threadIdx.x] = s;
    __syncthreads();
    for (int st = 128; st > 0; st >>= 1) {
        if (threadIdx.x < st) red[threadIdx.x] += red[threadIdx.x + st];
        __syncthreads();
    }
    if (threadIdx.x == 0) dout[0] = red[0] * (1.0f / (float)N_TOT);
}

// ---------------------------------------------------------------------------
extern "C" void kernel_launch(void* const* d_in, const int* in_sizes, int n_in,
                              void* d_out, int out_size, void* d_ws, size_t ws_size,
                              hipStream_t stream) {
    const float* feats = (const float*)d_in[0];   // [2048,2,128]
    const float* maskM = (const float*)d_in[1];   // [4096,4096]
    const float* Dm    = (const float*)d_in[2];   // [2048,128]
    const float* Lm    = (const float*)d_in[3];   // [2048,128]
    float* dout = (float*)d_out;                  // [1 + 4096*128]

    char* ws = (char*)d_ws;
    unsigned short* xbf  = (unsigned short*)ws;                              // 1 MiB
    float*          mrow = (float*)(ws + (1u << 20));                        // 16 KiB
    unsigned short* d2t  = (unsigned short*)(ws + (1u << 20) + (1u << 14));  // 1 MiB
    float*       rowloss = (float*)(ws + 2u * (1u << 20) + (1u << 14));      // 16 KiB

    k_prep_x  <<<512, 256, 0, stream>>>(feats, xbf, mrow);
    k_prep_d2t<<<2048, 256, 0, stream>>>(Dm, d2t);
    k_main    <<<256, 128, 0, stream>>>(xbf, d2t, mrow, maskM, Dm, Lm, dout, rowloss);
    k_loss    <<<1, 256, 0, stream>>>(rowloss, dout);
}
